// ThoughtsFormer_81776177316305
// MI455X (gfx1250) — compile-verified, matrix operand order checked by reading, not run on hardware
//
#include <hip/hip_runtime.h>
#include <hip/hip_bf16.h>
#include <math.h>

// Problem constants (match reference module constants).
#define BB    2
#define SS    1024
#define EE    768
#define HHD   8
#define LL    8
#define FFD   2048
#define VV    32000
#define NTT   1
#define REALC 512

// LDS B-tile row stride in halves: 32 data + 8 pad = 80 bytes
// (16B aligned for ds_load_b128, spreads rows across banks).
#define LDB_STRIDE 40

typedef __attribute__((ext_vector_type(16))) __bf16 v16bf;
typedef __attribute__((ext_vector_type(8)))  __bf16 v8bf;
typedef __attribute__((ext_vector_type(8)))  float  v8f;

__device__ __forceinline__ float gelu_exact(float x) {
    return 0.5f * x * (1.0f + erff(x * 0.70710678118654752f));
}

// Async global->LDS copy of 16B (gfx1250 GLOBAL_LOAD_ASYNC_TO_LDS_B128).
// INST_OFFSET is added to BOTH the global and the LDS address (ISA 15.18.3),
// so one VGPR pair + one LDS-addr VGPR serves two 16B beats.
__device__ __forceinline__ void async_b128(unsigned lds_addr, const void* gptr) {
    asm volatile("global_load_async_to_lds_b128 %0, %1, off"
                 :: "v"(lds_addr), "v"(gptr) : "memory");
}
__device__ __forceinline__ void async_b128_off16(unsigned lds_addr, const void* gptr) {
    asm volatile("global_load_async_to_lds_b128 %0, %1, off offset:16"
                 :: "v"(lds_addr), "v"(gptr) : "memory");
}
__device__ __forceinline__ void wait_async0() {
    asm volatile("s_wait_asynccnt 0x0" ::: "memory");
}

// A-fragment loader from row-major global fp32 (K contiguous).
// ISA 16-bit 16x32 A layout: lane holds row (lane&15); lanes<16 carry K in
// {0..7, 16..23}, lanes>=16 carry K in {8..15, 24..31} relative to k0.
__device__ __forceinline__ v16bf frag_rows(const float* __restrict__ base, int ld,
                                           int r, int k0, int lane) {
    const float* p = base + (size_t)r * ld + k0 + ((lane & 16) ? 8 : 0);
    v16bf f;
#pragma unroll
    for (int e = 0; e < 8; ++e) f[e]     = (__bf16)p[e];
#pragma unroll
    for (int e = 0; e < 8; ++e) f[8 + e] = (__bf16)p[16 + e];
    return f;
}

// B-fragment loader from an LDS-staged bf16 tile row (two 16B ds loads).
__device__ __forceinline__ v16bf frag_lds(const __bf16* p) {
    const v8bf lo = *(const v8bf*)(p);
    const v8bf hi = *(const v8bf*)(p + 16);
    v16bf f;
#pragma unroll
    for (int e = 0; e < 8; ++e) { f[e] = lo[e]; f[8 + e] = hi[e]; }
    return f;
}

// C = act(A[M,K] @ op(B) + bias), bf16 WMMA, f32 accumulate.
// BMODE 0: B = W[N,K] fp32 row-major (C = A @ W^T), convert while staging.
// BMODE 1: B = B[K,N] fp32 row-major (C = A @ B), transpose while staging.
// BMODE 2: B = W[N,K] bf16 row-major, staged via async global->LDS b128 copies
//          (no VALU in the staging path; ASYNCcnt tracked). NTILES must be 4.
// Block = 128 threads = 4 waves. Block tile: 128 rows x (NTILES*16) cols.
// Wave w owns rows [m0 + 32w, +32): 2 A-frags reused over NTILES B-frags.
// Requirements: M % 128 == 0, N % (NTILES*16) == 0, K % 32 == 0.
template <int ACT, int NTILES, int BMODE>
__global__ void __launch_bounds__(128)
wmma_gemm(const float* __restrict__ A, const void* __restrict__ Bv,
          const float* __restrict__ bias, float* __restrict__ C,
          int K, int lda, int ldb, int ldc) {
    constexpr int BROWS = NTILES * 16;
    static_assert(BMODE != 2 || NTILES == 4, "async staging assumes 64-row tiles");
    __shared__ __bf16 ldsB[BROWS * LDB_STRIDE];

    const int lane = threadIdx.x & 31;
    const int wave = threadIdx.x >> 5;
    const int m0 = blockIdx.y * 128 + wave * 32;
    const int n0 = blockIdx.x * BROWS;

    v8f acc[2][NTILES];
#pragma unroll
    for (int rf = 0; rf < 2; ++rf)
#pragma unroll
        for (int t = 0; t < NTILES; ++t)
#pragma unroll
            for (int v = 0; v < 8; ++v) acc[rf][t][v] = 0.0f;

    const int klo   = (lane & 16) ? 8 : 0;
    const int arow0 = m0 + (lane & 15);

    for (int k0 = 0; k0 < K; k0 += 32) {
        // ---- stage B tile (BROWS x 32) into LDS as bf16 ----
        if constexpr (BMODE == 2) {
            // rows of bf16 weights: 64B contiguous; 2 async b128 per thread
            const __bf16* Wb = (const __bf16*)Bv;
            const int r    = (int)(threadIdx.x >> 1);   // 0..63
            const int half = (int)(threadIdx.x & 1);    // 16 halves each
            const __bf16* src = Wb + (size_t)(n0 + r) * ldb + k0 + half * 16;
            unsigned dst = (unsigned)(uintptr_t)(ldsB + r * LDB_STRIDE + half * 16);
            async_b128(dst, src);
            async_b128_off16(dst, src);
            wait_async0();             // this wave's rows are in LDS
        } else if constexpr (BMODE == 1) {
            const float* Bm = (const float*)Bv;
            for (int n = threadIdx.x; n < BROWS; n += 128) {
                const float* src = Bm + (size_t)k0 * ldb + (n0 + n);
                __bf16* dst = ldsB + n * LDB_STRIDE;
#pragma unroll
                for (int k = 0; k < 32; ++k) dst[k] = (__bf16)src[(size_t)k * ldb];
            }
            if (k0 + 32 < K)
                __builtin_prefetch(Bm + (size_t)(k0 + 32) * ldb + n0 +
                                   (int)(threadIdx.x % BROWS), 0, 1);
        } else {
            const float* Bm = (const float*)Bv;
            const int half = threadIdx.x & 1;
            for (int r = (int)(threadIdx.x >> 1); r < BROWS; r += 64) {
                const float* src = Bm + (size_t)(n0 + r) * ldb + k0 + half * 16;
                __bf16* dst = ldsB + r * LDB_STRIDE + half * 16;
#pragma unroll
                for (int e = 0; e < 16; ++e) dst[e] = (__bf16)src[e];
            }
            if (k0 + 32 < K)
                __builtin_prefetch(Bm + (size_t)(n0 + (int)(threadIdx.x >> 1)) * ldb +
                                   (k0 + 32) + half * 16, 0, 1);
        }
        __syncthreads();

        const v16bf a0 = frag_rows(A, lda, arow0,      k0, lane);
        const v16bf a1 = frag_rows(A, lda, arow0 + 16, k0, lane);

#pragma unroll
        for (int t = 0; t < NTILES; ++t) {
            const __bf16* p = ldsB + (t * 16 + (lane & 15)) * LDB_STRIDE + klo;
            const v16bf bfrag = frag_lds(p);
            acc[0][t] = __builtin_amdgcn_wmma_f32_16x16x32_bf16(
                false, a0, false, bfrag, (short)0, acc[0][t], false, false);
            acc[1][t] = __builtin_amdgcn_wmma_f32_16x16x32_bf16(
                false, a1, false, bfrag, (short)0, acc[1][t], false, false);
        }
        __syncthreads();
    }

    // C layout: element v in lane L -> row base + v + (L>=16 ? 8 : 0), col (L&15)
    const int rhi = (lane & 16) ? 8 : 0;
#pragma unroll
    for (int rf = 0; rf < 2; ++rf) {
#pragma unroll
        for (int t = 0; t < NTILES; ++t) {
            const int n = n0 + t * 16 + (lane & 15);
            const float bv = bias ? bias[n] : 0.0f;
#pragma unroll
            for (int v = 0; v < 8; ++v) {
                const int m = m0 + rf * 16 + rhi + v;
                float r = acc[rf][t][v] + bv;
                if (ACT == 1) r = gelu_exact(r);
                C[(size_t)m * ldc + n] = r;
            }
        }
    }
}

// fp32 -> bf16 weight pre-conversion (amortized: weights read once per pass).
__global__ void cvt_bf16_kernel(const float* __restrict__ src, __bf16* __restrict__ dst, int n) {
    int i = blockIdx.x * blockDim.x + threadIdx.x;
    int i4 = i * 4;
    if (i4 + 3 < n) {
#pragma unroll
        for (int j = 0; j < 4; ++j) dst[i4 + j] = (__bf16)src[i4 + j];
    } else {
        for (int j = i4; j < n; ++j) dst[j] = (__bf16)src[j];
    }
}

// h = x + sinusoidal PE(token pos) + thought-position embedding (zero-pad tail).
__global__ void embed_kernel(const float* __restrict__ x, const float* __restrict__ tpe,
                             float* __restrict__ h, int total, int bs, int realbs) {
    int idx = blockIdx.x * blockDim.x + threadIdx.x;
    if (idx >= total) return;
    int e = idx % EE;
    int t = idx / EE;
    int s = t % SS;
    float v;
    if (s < realbs) {
        int pos = s / bs;
        int th  = s % bs;
        int pair = e >> 1;
        float dv  = expf((float)(2 * pair) * (-logf(10000.0f) / (float)EE));
        float ang = (float)pos * dv;
        float pe  = (e & 1) ? cosf(ang) : sinf(ang);
        v = x[idx] + pe + tpe[th * EE + e];
    } else {
        v = 0.0f;
    }
    h[idx] = v;
}

// In-place softmax over keys; bias is +1.0 where M[k,q] is false (per reference).
__global__ void __launch_bounds__(256)
softmax_kernel(float* __restrict__ scores, int bs, int realbs, float scale) {
    const int q  = blockIdx.x;
    const int bh = blockIdx.y;
    float* row = scores + ((size_t)bh * SS + q) * SS;
    __shared__ float red[256];
    const int tid = threadIdx.x;
    const int tokq = q / bs, pq = q % bs;

    float mx = -3.4e38f;
    for (int k = tid; k < SS; k += 256) {
        int tokk = k / bs, pk = k % bs;
        bool mkq = (tokk == tokq && pq >= pk) || (pk == 0 && q > k && q < realbs);
        float sv = row[k] * scale + (mkq ? 0.0f : 1.0f);
        mx = fmaxf(mx, sv);
    }
    red[tid] = mx; __syncthreads();
    for (int st = 128; st > 0; st >>= 1) {
        if (tid < st) red[tid] = fmaxf(red[tid], red[tid + st]);
        __syncthreads();
    }
    mx = red[0]; __syncthreads();

    float sum = 0.0f;
    for (int k = tid; k < SS; k += 256) {
        int tokk = k / bs, pk = k % bs;
        bool mkq = (tokk == tokq && pq >= pk) || (pk == 0 && q > k && q < realbs);
        float sv = row[k] * scale + (mkq ? 0.0f : 1.0f);
        float ev = expf(sv - mx);
        row[k] = ev;
        sum += ev;
    }
    red[tid] = sum; __syncthreads();
    for (int st = 128; st > 0; st >>= 1) {
        if (tid < st) red[tid] += red[tid + st];
        __syncthreads();
    }
    const float inv = 1.0f / red[0];
    for (int k = tid; k < SS; k += 256) row[k] *= inv;
}

// out = LN(res + delta) * g + b, optional nan_to_num; delta may be null.
__global__ void __launch_bounds__(256)
ln_kernel(const float* __restrict__ res, const float* __restrict__ delta,
          const float* __restrict__ g, const float* __restrict__ b,
          float* __restrict__ out, int donan) {
    const int t = blockIdx.x;
    const int tid = threadIdx.x;
    const float* r = res + (size_t)t * EE;
    const float* d = delta ? delta + (size_t)t * EE : nullptr;
    __shared__ float red[256];
    __shared__ float s_mean, s_inv;

    float s = 0.0f;
    for (int e = tid; e < EE; e += 256) s += r[e] + (d ? d[e] : 0.0f);
    red[tid] = s; __syncthreads();
    for (int st = 128; st > 0; st >>= 1) {
        if (tid < st) red[tid] += red[tid + st];
        __syncthreads();
    }
    if (tid == 0) s_mean = red[0] / (float)EE;
    __syncthreads();
    const float mean = s_mean;

    float vs = 0.0f;
    for (int e = tid; e < EE; e += 256) {
        float v = r[e] + (d ? d[e] : 0.0f) - mean;
        vs += v * v;
    }
    red[tid] = vs; __syncthreads();
    for (int st = 128; st > 0; st >>= 1) {
        if (tid < st) red[tid] += red[tid + st];
        __syncthreads();
    }
    if (tid == 0) s_inv = rsqrtf(red[0] / (float)EE + 1e-5f);
    __syncthreads();
    const float inv = s_inv;

    for (int e = tid; e < EE; e += 256) {
        float v = (r[e] + (d ? d[e] : 0.0f) - mean) * inv * g[e] + b[e];
        if (donan) {
            if (isnan(v)) v = 0.0f;
            else if (isinf(v)) v = copysignf(3.4028235e38f, v);
        }
        out[(size_t)t * EE + e] = v;
    }
}

// a[b, r, :] = h[b, r*bs + NT, :]
__global__ void gather_kernel(const float* __restrict__ h, float* __restrict__ a, int bs) {
    int idx = blockIdx.x * blockDim.x + threadIdx.x;
    const int total = BB * REALC * EE;
    if (idx >= total) return;
    int e = idx % EE;
    int t = idx / EE;
    int r = t % REALC;
    int b = t / REALC;
    int srow = r * bs + (bs - 1);
    a[idx] = h[((size_t)(b * SS + srow)) * EE + e];
}

// value[t] = dot(vh[t,:], Wv2) + bv2  (N=1 head, not worth WMMA)
__global__ void __launch_bounds__(256)
value_kernel(const float* __restrict__ vh, const float* __restrict__ Wv2,
             const float* __restrict__ bv2, float* __restrict__ out) {
    const int t = blockIdx.x;
    const int tid = threadIdx.x;
    __shared__ float red[256];
    float s = 0.0f;
    const float* p = vh + (size_t)t * FFD;
    for (int f = tid; f < FFD; f += 256) s += p[f] * Wv2[f];
    red[tid] = s; __syncthreads();
    for (int st = 128; st > 0; st >>= 1) {
        if (tid < st) red[tid] += red[tid + st];
        __syncthreads();
    }
    if (tid == 0) out[t] = red[0] + bv2[0];
}

extern "C" void kernel_launch(void* const* d_in, const int* in_sizes, int n_in,
                              void* d_out, int out_size, void* d_ws, size_t ws_size,
                              hipStream_t stream) {
    (void)in_sizes; (void)n_in; (void)out_size; (void)ws_size;

    const float* x    = (const float*)d_in[0];
    const float* tpe  = (const float*)d_in[1];
    const float* Wqkv = (const float*)d_in[2];
    const float* bqkv = (const float*)d_in[3];
    const float* W1   = (const float*)d_in[4];
    const float* b1   = (const float*)d_in[5];
    const float* W2   = (const float*)d_in[6];
    const float* b2   = (const float*)d_in[7];
    const float* g1   = (const float*)d_in[8];
    const float* be1  = (const float*)d_in[9];
    const float* g2   = (const float*)d_in[10];
    const float* be2  = (const float*)d_in[11];
    const float* gf   = (const float*)d_in[12];
    const float* bfp  = (const float*)d_in[13];
    const float* Wp1  = (const float*)d_in[14];
    const float* bp1  = (const float*)d_in[15];
    const float* Wp2  = (const float*)d_in[16];
    const float* bp2  = (const float*)d_in[17];
    const float* Wv1  = (const float*)d_in[18];
    const float* bv1  = (const float*)d_in[19];
    const float* Wv2  = (const float*)d_in[20];
    const float* bv2  = (const float*)d_in[21];

    float* out = (float*)d_out;

    const int bs = NTT + 1;             // 2
    const int realbs = REALC * bs;      // 1024 == S (no padding)
    const int dh = EE / HHD;            // 96
    const float scale = 1.0f / sqrtf((float)dh);

    // Workspace slabs: fp32 activations, then bf16 weight shadow copies.
    float* ws = (float*)d_ws;
    size_t off = 0;
    float* h   = ws + off; off += (size_t)BB * SS * EE;        // hidden state
    float* qkv = ws + off; off += (size_t)BB * SS * 3 * EE;    // fused qkv
    float* sc  = ws + off; off += (size_t)BB * HHD * SS * SS;  // scores / probs
    float* o   = ws + off; off += (size_t)BB * SS * EE;        // attn out / ffn out
    float* t1  = ws + off; off += (size_t)BB * SS * FFD;       // ffn hidden
    float* a   = ws + off; off += (size_t)BB * REALC * EE;     // gathered acts
    float* ph  = ws + off; off += (size_t)BB * REALC * FFD;    // head hidden

    __bf16* wb = (__bf16*)(ws + off);
    size_t boff = 0;
    __bf16* Wqkv_b = wb + boff; boff += (size_t)LL * 3 * EE * EE;
    __bf16* W1_b   = wb + boff; boff += (size_t)LL * FFD * EE;
    __bf16* W2_b   = wb + boff; boff += (size_t)LL * EE * FFD;
    __bf16* Wp1_b  = wb + boff; boff += (size_t)FFD * EE;
    __bf16* Wp2_b  = wb + boff; boff += (size_t)VV * FFD;
    __bf16* Wv1_b  = wb + boff; boff += (size_t)FFD * EE;

    // Pre-convert weights to bf16 (read once; all GEMM B-traffic halves).
    {
        struct { const float* s; __bf16* d; int n; } cv[6] = {
            { Wqkv, Wqkv_b, LL * 3 * EE * EE },
            { W1,   W1_b,   LL * FFD * EE },
            { W2,   W2_b,   LL * EE * FFD },
            { Wp1,  Wp1_b,  FFD * EE },
            { Wp2,  Wp2_b,  VV * FFD },
            { Wv1,  Wv1_b,  FFD * EE },
        };
        for (int i = 0; i < 6; ++i) {
            int quads = (cv[i].n + 3) / 4;
            cvt_bf16_kernel<<<(quads + 255) / 256, 256, 0, stream>>>(cv[i].s, cv[i].d, cv[i].n);
        }
    }

    const int Mtok = BB * SS;   // 2048 token rows
    const dim3 blk(128);

    {
        int total = BB * SS * EE;
        embed_kernel<<<(total + 255) / 256, 256, 0, stream>>>(x, tpe, h, total, bs, realbs);
    }

    for (int l = 0; l < LL; ++l) {
        const __bf16* Wq = Wqkv_b + (size_t)l * 3 * EE * EE;
        const float*  bq = bqkv + (size_t)l * 3 * EE;

        // qkv = h @ Wq^T + bq   (2048 x 2304 x 768), async bf16 B staging
        wmma_gemm<0, 4, 2><<<dim3((3 * EE) / 64, Mtok / 128), blk, 0, stream>>>(
            h, Wq, bq, qkv, EE, EE, EE, 3 * EE);

        // scores = q @ k^T per (b, head)   (1024 x 1024 x 96), fp32 B
        for (int b = 0; b < BB; ++b)
            for (int hh = 0; hh < HHD; ++hh) {
                const float* qp = qkv + (size_t)b * SS * 3 * EE + hh * dh;
                const float* kp = qp + EE;
                float* sp = sc + ((size_t)(b * HHD + hh)) * SS * SS;
                wmma_gemm<0, 4, 0><<<dim3(SS / 64, SS / 128), blk, 0, stream>>>(
                    qp, kp, nullptr, sp, dh, 3 * EE, 3 * EE, SS);
            }

        softmax_kernel<<<dim3(SS, BB * HHD), 256, 0, stream>>>(sc, bs, realbs, scale);

        // o = P @ V per (b, head)   (1024 x 96 x 1024), B is [K,N]-major fp32
        for (int b = 0; b < BB; ++b)
            for (int hh = 0; hh < HHD; ++hh) {
                const float* vp = qkv + (size_t)b * SS * 3 * EE + 2 * EE + hh * dh;
                const float* sp = sc + ((size_t)(b * HHD + hh)) * SS * SS;
                float* op = o + (size_t)b * SS * EE + hh * dh;
                wmma_gemm<0, 6, 1><<<dim3(1, SS / 128), blk, 0, stream>>>(
                    sp, vp, nullptr, op, SS, SS, 3 * EE, EE);
            }

        // h = LN(h + o)
        ln_kernel<<<Mtok, 256, 0, stream>>>(h, o, g1 + (size_t)l * EE, be1 + (size_t)l * EE, h, 0);

        // t1 = gelu(h @ W1^T + b1)   (2048 x 2048 x 768)
        wmma_gemm<1, 4, 2><<<dim3(FFD / 64, Mtok / 128), blk, 0, stream>>>(
            h, W1_b + (size_t)l * FFD * EE, b1 + (size_t)l * FFD, t1, EE, EE, EE, FFD);

        // o = t1 @ W2^T + b2   (2048 x 768 x 2048)
        wmma_gemm<0, 4, 2><<<dim3(EE / 64, Mtok / 128), blk, 0, stream>>>(
            t1, W2_b + (size_t)l * EE * FFD, b2 + (size_t)l * EE, o, FFD, FFD, FFD, EE);

        // h = nan_to_num(LN(h + o))
        ln_kernel<<<Mtok, 256, 0, stream>>>(h, o, g2 + (size_t)l * EE, be2 + (size_t)l * EE, h, 1);
    }

    // final LN
    ln_kernel<<<Mtok, 256, 0, stream>>>(h, nullptr, gf, bfp, h, 0);

    {
        int total = BB * REALC * EE;
        gather_kernel<<<(total + 255) / 256, 256, 0, stream>>>(h, a, bs);
    }

    const int Ma = BB * REALC;  // 1024
    // policy head
    wmma_gemm<1, 4, 2><<<dim3(FFD / 64, Ma / 128), blk, 0, stream>>>(
        a, Wp1_b, bp1, ph, EE, EE, EE, FFD);
    wmma_gemm<0, 4, 2><<<dim3(VV / 64, Ma / 128), blk, 0, stream>>>(
        ph, Wp2_b, bp2, out, FFD, FFD, FFD, VV);
    // value head (reuse ph after policy GEMM consumed it; same stream -> ordered)
    wmma_gemm<1, 4, 2><<<dim3(FFD / 64, Ma / 128), blk, 0, stream>>>(
        a, Wv1_b, bv1, ph, EE, EE, EE, FFD);
    value_kernel<<<Ma, 256, 0, stream>>>(ph, Wv2, bv2, out + (size_t)Ma * VV);
}
